// FusionAttention_23708219474123
// MI455X (gfx1250) — compile-verified, matrix operand order checked by reading, not run on hardware
//
#include <hip/hip_runtime.h>

// ---------------------------------------------------------------------------
// CDNA5 (gfx1250, wave32) fused dual-stream attention, BF16-resident pipeline.
//  - all GEMM operands pre-converted to bf16 (weights pre-transposed to [NxK])
//  - staging via GLOBAL_LOAD_ASYNC_TO_LDS_B128 (ASYNCcnt), double-buffered LDS
//  - matrix math via v_wmma_f32_16x16x32_bf16, f32 accumulation
//  - softmax + P·V fused into one kernel (online row stats, in-place normalize)
// ---------------------------------------------------------------------------

typedef __attribute__((ext_vector_type(4)))  __bf16 v4bf;
typedef __attribute__((ext_vector_type(8)))  __bf16 v8bf;
typedef __attribute__((ext_vector_type(16))) __bf16 v16bf;
typedef __attribute__((ext_vector_type(8)))  float  v8f;

#define BM 128
#define BN 64
#define BK 32
#define AS_S 40   // LDS row stride (32+8 pad) in bf16 elems; 80B keeps 16B align
#define BS_S 40

__device__ __forceinline__ unsigned ldsOffsetOf(const void* p) {
  // generic LDS pointers carry the allocation-relative offset in the low 32 bits
  return (unsigned)(unsigned long long)p;
}

__device__ __forceinline__ void async_b128(const void* gaddr, unsigned ldsOff) {
  // GLOBAL_LOAD_ASYNC_TO_LDS_B128: vdst = per-lane LDS byte offset, vaddr = 64b
  asm volatile("global_load_async_to_lds_b128 %0, %1, off"
               :: "v"(ldsOff), "v"(gaddr) : "memory");
}

// ---------------------------------------------------------------------------
// Generic bf16 GEMM: C[m,n] = alpha * sum_k A[m,k]*B[n,k] (+ Cin if accumulate)
// A: bf16 [MxK]; B: bf16 [NxK] row-major (pre-transposed weights / K rows)
// CMODE: 0 = f32 out, 1 = bf16 out (optional bf16 accumulate), 2 = bf16 V^T
//        layout [b,h][dk][s] hard-wired for S=1024, DK=64, H=8.
// ---------------------------------------------------------------------------
template <int CMODE>
__global__ __launch_bounds__(256)
void gemm_wmma(const __bf16* __restrict__ Aop, const __bf16* __restrict__ Bop,
               const void* Cin_, void* Cout_,
               int K, int lda, int ldb, int ldc,
               long long sAb, long long sAh,
               long long sBb, long long sBh,
               long long sCb, long long sCh,
               int nH, float alpha, int accumulate)
{
  __shared__ unsigned short As[2][BM * AS_S];   // [m][k] bf16, double buffered
  __shared__ unsigned short Bs[2][BN * BS_S];   // [n][k] bf16, double buffered

  const int z  = blockIdx.z;
  const int zb = z / nH;
  const int zh = z - zb * nH;
  const __bf16* A = Aop + zb * sAb + zh * sAh;
  const __bf16* B = Bop + zb * sBb + zh * sBh;
  const long long cofs = zb * sCb + zh * sCh;

  const int m0 = blockIdx.y * BM;
  const int n0 = blockIdx.x * BN;

  const int tid   = threadIdx.x;
  const int lane  = tid & 31;
  const int wv    = tid >> 5;       // 8 waves: 4 (M) x 2 (N)
  const int waveM = wv >> 1;
  const int waveN = wv & 1;
  const int hk    = lane >> 4;      // K-block select per ISA 16-bit layout
  const int l16   = lane & 15;
  const int kb    = hk << 3;

  v8f acc[2][2];
  #pragma unroll
  for (int i = 0; i < 2; ++i)
    #pragma unroll
    for (int j = 0; j < 2; ++j) acc[i][j] = {};

  auto stageA = [&](int buf, int k0) {          // 2 async b128 per thread
    #pragma unroll
    for (int it = 0; it < 2; ++it) {
      const int flat = tid + (it << 8);          // 512 x 16B chunks
      const int row  = flat >> 2;
      const int c8   = (flat & 3) << 3;
      async_b128(A + (long long)(m0 + row) * lda + k0 + c8,
                 ldsOffsetOf(&As[buf][row * AS_S + c8]));
    }
  };
  auto stageB = [&](int buf, int k0) {          // 1 async b128 per thread
    const int row = tid >> 2;
    const int c8  = (tid & 3) << 3;
    async_b128(B + (long long)(n0 + row) * ldb + k0 + c8,
               ldsOffsetOf(&Bs[buf][row * BS_S + c8]));
  };

  stageA(0, 0);
  stageB(0, 0);

  int buf = 0;
  for (int k0 = 0; k0 < K; k0 += BK) {
    if (k0 + BK < K) {
      stageA(buf ^ 1, k0 + BK);
      stageB(buf ^ 1, k0 + BK);
      asm volatile("s_wait_asynccnt 0x3" ::: "memory");  // prev stage done
    } else {
      asm volatile("s_wait_asynccnt 0x0" ::: "memory");
    }
    __syncthreads();

    v16bf af[2], bfv[2];
    #pragma unroll
    for (int i = 0; i < 2; ++i) {
      const int m = waveM * 32 + i * 16 + l16;
      const v8bf lo = *(const v8bf*)&As[buf][m * AS_S + kb];
      const v8bf hi = *(const v8bf*)&As[buf][m * AS_S + kb + 16];
      af[i] = __builtin_shufflevector(lo, hi, 0,1,2,3,4,5,6,7,8,9,10,11,12,13,14,15);
    }
    #pragma unroll
    for (int j = 0; j < 2; ++j) {
      const int n = waveN * 32 + j * 16 + l16;
      const v8bf lo = *(const v8bf*)&Bs[buf][n * BS_S + kb];
      const v8bf hi = *(const v8bf*)&Bs[buf][n * BS_S + kb + 16];
      bfv[j] = __builtin_shufflevector(lo, hi, 0,1,2,3,4,5,6,7,8,9,10,11,12,13,14,15);
    }
    #pragma unroll
    for (int i = 0; i < 2; ++i)
      #pragma unroll
      for (int j = 0; j < 2; ++j)
        acc[i][j] = __builtin_amdgcn_wmma_f32_16x16x32_bf16(
            false, af[i], false, bfv[j], (short)0, acc[i][j], false, false);

    __syncthreads();
    buf ^= 1;
  }

  #pragma unroll
  for (int i = 0; i < 2; ++i) {
    #pragma unroll
    for (int j = 0; j < 2; ++j) {
      const int gm = m0 + waveM * 32 + i * 16 + hk * 8;
      const int gn = n0 + waveN * 32 + j * 16 + l16;
      #pragma unroll
      for (int r = 0; r < 8; ++r) {
        const long long gmr = gm + r;
        float v = alpha * acc[i][j][r];
        if constexpr (CMODE == 0) {
          ((float*)Cout_)[cofs + gmr * ldc + gn] = v;
        } else if constexpr (CMODE == 1) {
          const long long idx = cofs + gmr * ldc + gn;
          if (accumulate) v += (float)((const __bf16*)Cin_)[idx];
          ((__bf16*)Cout_)[idx] = (__bf16)v;
        } else {            // V^T: [b,h][dk][s]
          const int b = (int)(gmr >> 10), kk = (int)(gmr & 1023);
          const int h = gn >> 6,          d  = gn & 63;
          ((__bf16*)Cout_)[(((long long)((b << 3) + h)) << 16) + (d << 10) + kk] =
              (__bf16)v;
        }
      }
    }
  }
}

// ---------------------------------------------------------------------------
// Fused softmax + P·V per (b,h, 128-row slab).
//  Sc holds raw scaled logits (written by the QK^T GEMM); this kernel
//  pass 1: online row max / sum over the 128x1024 slab (flash-style),
//  pass 2: normalize in place (final probabilities -> d_out), feed bf16 P
//          tiles into WMMA against async-staged V^T rows, write O (bf16).
// ---------------------------------------------------------------------------
__global__ __launch_bounds__(256)
void attn_pv_fused(float* __restrict__ Sc, const __bf16* __restrict__ vT,
                   __bf16* __restrict__ O)
{
  __shared__ unsigned short As[2][BM * AS_S];
  __shared__ unsigned short Bs[2][BN * BS_S];
  __shared__ float rmax[BM];
  __shared__ float rinv[BM];

  const int z  = blockIdx.z;                 // b*8 + h
  const int zb = z >> 3, zh = z & 7;
  float*        S  = Sc + (long long)z * (1024LL * 1024);
  const __bf16* B  = vT + (long long)z * (64LL * 1024);
  __bf16*       Ob = O + (long long)zb * (1024LL * 512) + zh * 64;

  const int m0  = blockIdx.y * BM;
  const int tid = threadIdx.x;
  const int lane = tid & 31, wv = tid >> 5;
  const int waveM = wv >> 1, waveN = wv & 1;
  const int hk = lane >> 4, l16 = lane & 15, kb = hk << 3;

  // ---- pass 1: online row statistics (8 lanes cooperate per row) ----------
  const int prow = tid >> 3;                 // row group (same for 8 lanes)
  const int pcol = (tid & 7) << 2;           // float4 column base
  float m_run[4], s_run[4];
  #pragma unroll
  for (int it = 0; it < 4; ++it) { m_run[it] = -3.0e38f; s_run[it] = 0.0f; }

  for (int k0 = 0; k0 < 1024; k0 += BK) {
    #pragma unroll
    for (int it = 0; it < 4; ++it) {
      const int row = prow + (it << 5);
      const float* g = S + (long long)(m0 + row) * 1024 + k0 + pcol;
      const float4 f = *(const float4*)g;
      if (k0 + BK < 1024) __builtin_prefetch(g + BK, 0, 1);
      float cm = fmaxf(fmaxf(f.x, f.y), fmaxf(f.z, f.w));
      cm = fmaxf(cm, __shfl_xor(cm, 1, 32));
      cm = fmaxf(cm, __shfl_xor(cm, 2, 32));
      cm = fmaxf(cm, __shfl_xor(cm, 4, 32));
      float ce = __expf(f.x - cm) + __expf(f.y - cm) +
                 __expf(f.z - cm) + __expf(f.w - cm);
      ce += __shfl_xor(ce, 1, 32);
      ce += __shfl_xor(ce, 2, 32);
      ce += __shfl_xor(ce, 4, 32);
      const float mo = m_run[it];
      const float mn = fmaxf(mo, cm);
      s_run[it] = s_run[it] * __expf(mo - mn) + ce * __expf(cm - mn);
      m_run[it] = mn;
    }
  }
  if ((tid & 7) == 0) {
    #pragma unroll
    for (int it = 0; it < 4; ++it) {
      rmax[prow + (it << 5)] = m_run[it];
      rinv[prow + (it << 5)] = 1.0f / s_run[it];
    }
  }
  __syncthreads();

  // ---- pass 2: normalize + write P + P·V ----------------------------------
  v8f acc[2][2];
  #pragma unroll
  for (int i = 0; i < 2; ++i)
    #pragma unroll
    for (int j = 0; j < 2; ++j) acc[i][j] = {};

  auto stageA = [&](int buf, int k0) {       // slab re-read (L2-hot), exp+store
    #pragma unroll
    for (int it = 0; it < 4; ++it) {
      const int flat = tid + (it << 8);
      const int row  = flat >> 3;
      const int c4   = (flat & 7) << 2;
      float* g = S + (long long)(m0 + row) * 1024 + k0 + c4;
      float4 f = *(float4*)g;
      const float mr = rmax[row], ir = rinv[row];
      f.x = __expf(f.x - mr) * ir;  f.y = __expf(f.y - mr) * ir;
      f.z = __expf(f.z - mr) * ir;  f.w = __expf(f.w - mr) * ir;
      *(float4*)g = f;                       // final normalized score output
      v4bf p = { (__bf16)f.x, (__bf16)f.y, (__bf16)f.z, (__bf16)f.w };
      *(v4bf*)&As[buf][row * AS_S + c4] = p;
    }
  };
  auto stageB = [&](int buf, int k0) {       // V^T rows: 1 async b128/thread
    const int row = tid >> 2;
    const int c8  = (tid & 3) << 3;
    async_b128(B + (long long)row * 1024 + k0 + c8,
               ldsOffsetOf(&Bs[buf][row * BS_S + c8]));
  };

  stageA(0, 0);
  stageB(0, 0);

  int buf = 0;
  for (int k0 = 0; k0 < 1024; k0 += BK) {
    if (k0 + BK < 1024) {
      stageA(buf ^ 1, k0 + BK);
      stageB(buf ^ 1, k0 + BK);
      asm volatile("s_wait_asynccnt 0x1" ::: "memory");
    } else {
      asm volatile("s_wait_asynccnt 0x0" ::: "memory");
    }
    __syncthreads();

    v16bf af[2], bfv[2];
    #pragma unroll
    for (int i = 0; i < 2; ++i) {
      const int m = waveM * 32 + i * 16 + l16;
      const v8bf lo = *(const v8bf*)&As[buf][m * AS_S + kb];
      const v8bf hi = *(const v8bf*)&As[buf][m * AS_S + kb + 16];
      af[i] = __builtin_shufflevector(lo, hi, 0,1,2,3,4,5,6,7,8,9,10,11,12,13,14,15);
    }
    #pragma unroll
    for (int j = 0; j < 2; ++j) {
      const int n = waveN * 32 + j * 16 + l16;
      const v8bf lo = *(const v8bf*)&Bs[buf][n * BS_S + kb];
      const v8bf hi = *(const v8bf*)&Bs[buf][n * BS_S + kb + 16];
      bfv[j] = __builtin_shufflevector(lo, hi, 0,1,2,3,4,5,6,7,8,9,10,11,12,13,14,15);
    }
    #pragma unroll
    for (int i = 0; i < 2; ++i)
      #pragma unroll
      for (int j = 0; j < 2; ++j)
        acc[i][j] = __builtin_amdgcn_wmma_f32_16x16x32_bf16(
            false, af[i], false, bfv[j], (short)0, acc[i][j], false, false);

    __syncthreads();
    buf ^= 1;
  }

  #pragma unroll
  for (int i = 0; i < 2; ++i) {
    #pragma unroll
    for (int j = 0; j < 2; ++j) {
      const int gm = m0 + waveM * 32 + i * 16 + hk * 8;
      const int gn = waveN * 32 + j * 16 + l16;   // n0 == 0, N == 64
      #pragma unroll
      for (int r = 0; r < 8; ++r)
        Ob[(long long)(gm + r) * 512 + gn] = (__bf16)acc[i][j][r];
    }
  }
}

// elementwise f32 -> bf16 (vectorized)
__global__ __launch_bounds__(256)
void cvt_bf16(const float* __restrict__ in, __bf16* __restrict__ out)
{
  const long long i = (long long)blockIdx.x * 256 + threadIdx.x;
  const float4 f = ((const float4*)in)[i];
  v4bf p = { (__bf16)f.x, (__bf16)f.y, (__bf16)f.z, (__bf16)f.w };
  ((v4bf*)out)[i] = p;
}

// out[n*K + k] = (bf16) in[k*N + n]   (32x32 LDS tile transpose)
__global__ __launch_bounds__(256)
void transpose_cvt(const float* __restrict__ in, __bf16* __restrict__ out,
                   int K, int N)
{
  __shared__ float tl[32][33];
  const int k0 = blockIdx.y << 5, n0 = blockIdx.x << 5;
  const int tx = threadIdx.x & 31, ty = threadIdx.x >> 5;
  #pragma unroll
  for (int j = ty; j < 32; j += 8)
    tl[j][tx] = in[(long long)(k0 + j) * N + n0 + tx];
  __syncthreads();
  #pragma unroll
  for (int j = ty; j < 32; j += 8)
    out[(long long)(n0 + j) * K + k0 + tx] = (__bf16)tl[tx][j];
}

extern "C" void kernel_launch(void* const* d_in, const int* in_sizes, int n_in,
                              void* d_out, int out_size, void* d_ws, size_t ws_size,
                              hipStream_t stream) {
  (void)in_sizes; (void)n_in; (void)out_size; (void)ws_size;
  const float* a    = (const float*)d_in[0];
  const float* t    = (const float*)d_in[1];
  const float* W512[8] = {                       // 512x512 weights, dict order
    (const float*)d_in[2], (const float*)d_in[3], (const float*)d_in[4],
    (const float*)d_in[5], (const float*)d_in[6], (const float*)d_in[7],
    (const float*)d_in[8], (const float*)d_in[9] };
  const float* WF_t = (const float*)d_in[10];
  const float* WF_a = (const float*)d_in[11];

  float* out = (float*)d_out;

  const long long NE = 8LL * 1024 * 512;         // 4,194,304 elems
  __bf16* w = (__bf16*)d_ws;
  __bf16* abf = w;            __bf16* tbf = w + NE;
  __bf16* q_a = w + 2 * NE;   __bf16* k_a = w + 3 * NE;  __bf16* vT_a = w + 4 * NE;
  __bf16* q_t = w + 5 * NE;   __bf16* k_t = w + 6 * NE;  __bf16* vT_t = w + 7 * NE;
  __bf16* o_a = q_a;          __bf16* o_t = q_t;         // Q dead after scores
  __bf16* WT[8];
  for (int i = 0; i < 8; ++i) WT[i] = w + 8 * NE + (long long)i * 512 * 512;
  __bf16* WFtT = w + 8 * NE + 8LL * 512 * 512;
  __bf16* WFaT = WFtT + 64 * 64;
  __bf16* WQaT = WT[0]; __bf16* WKaT = WT[1]; __bf16* WVaT = WT[2]; __bf16* WOaT = WT[3];
  __bf16* WQtT = WT[4]; __bf16* WKtT = WT[5]; __bf16* WVtT = WT[6]; __bf16* WOtT = WT[7];

  float* out_a   = out;                          // [8,1024,512]
  float* out_t   = out + NE;
  float* score_a = out + 2 * NE;                 // [8,8,1024,1024]
  float* score_t = score_a + 64LL * 1024 * 1024;

  const dim3 blk(256);
  const long long SBH = 1024LL * 512;            // per-batch stride in [8192,512]
  const long long SSC = 1024LL * 1024;           // per-(b,h) score stride

  // 0) one-time bf16 conversion / weight transposition
  cvt_bf16<<<dim3(NE / 1024), blk, 0, stream>>>(a, abf);
  cvt_bf16<<<dim3(NE / 1024), blk, 0, stream>>>(t, tbf);
  {
    const dim3 g(16, 16);
    for (int i = 0; i < 8; ++i)
      transpose_cvt<<<g, blk, 0, stream>>>(W512[i], WT[i], 512, 512);
    const dim3 g2(2, 2);
    transpose_cvt<<<g2, blk, 0, stream>>>(WF_t, WFtT, 64, 64);
    transpose_cvt<<<g2, blk, 0, stream>>>(WF_a, WFaT, 64, 64);
  }

  // 1) QKV projections: [8192x512]·[512x512], bf16 out (V written as V^T)
  {
    const dim3 g(512 / BN, 8192 / BM, 1);
    gemm_wmma<1><<<g, blk, 0, stream>>>(abf, WQaT, q_a, q_a, 512, 512, 512, 512,
                                        0, 0, 0, 0, 0, 0, 1, 1.0f, 0);
    gemm_wmma<1><<<g, blk, 0, stream>>>(abf, WKaT, k_a, k_a, 512, 512, 512, 512,
                                        0, 0, 0, 0, 0, 0, 1, 1.0f, 0);
    gemm_wmma<2><<<g, blk, 0, stream>>>(abf, WVaT, vT_a, vT_a, 512, 512, 512, 512,
                                        0, 0, 0, 0, 0, 0, 1, 1.0f, 0);
    gemm_wmma<1><<<g, blk, 0, stream>>>(tbf, WQtT, q_t, q_t, 512, 512, 512, 512,
                                        0, 0, 0, 0, 0, 0, 1, 1.0f, 0);
    gemm_wmma<1><<<g, blk, 0, stream>>>(tbf, WKtT, k_t, k_t, 512, 512, 512, 512,
                                        0, 0, 0, 0, 0, 0, 1, 1.0f, 0);
    gemm_wmma<2><<<g, blk, 0, stream>>>(tbf, WVtT, vT_t, vT_t, 512, 512, 512, 512,
                                        0, 0, 0, 0, 0, 0, 1, 1.0f, 0);
  }

  // 2) sequential key fusion: k_a += k_t·WF_a ; then k_t += k_a·WF_t
  {
    const dim3 g(1, 8192 / BM, 8);
    gemm_wmma<1><<<g, blk, 0, stream>>>(k_t, WFaT, k_a, k_a, 64, 512, 64, 512,
                                        0, 64, 0, 0, 0, 64, 8, 1.0f, 1);
    gemm_wmma<1><<<g, blk, 0, stream>>>(k_a, WFtT, k_t, k_t, 64, 512, 64, 512,
                                        0, 64, 0, 0, 0, 64, 8, 1.0f, 1);
  }

  // 3) raw scaled logits = (Q·K^T)/8 per (b,h) -> d_out score regions
  {
    const dim3 g(1024 / BN, 1024 / BM, 64);
    gemm_wmma<0><<<g, blk, 0, stream>>>(q_a, k_a, score_a, score_a, 64,
                                        512, 512, 1024,
                                        SBH, 64, SBH, 64, 8 * SSC, SSC,
                                        8, 0.125f, 0);
    gemm_wmma<0><<<g, blk, 0, stream>>>(q_t, k_t, score_t, score_t, 64,
                                        512, 512, 1024,
                                        SBH, 64, SBH, 64, 8 * SSC, SSC,
                                        8, 0.125f, 0);
  }

  // 4) fused softmax + P·V: normalizes scores in place, writes O (bf16)
  {
    const dim3 g(1, 1024 / BM, 64);
    attn_pv_fused<<<g, blk, 0, stream>>>(score_a, vT_a, o_a);
    attn_pv_fused<<<g, blk, 0, stream>>>(score_t, vT_t, o_t);
  }

  // 5) output projections -> d_out (f32)
  {
    const dim3 g(512 / BN, 8192 / BM, 1);
    gemm_wmma<0><<<g, blk, 0, stream>>>(o_a, WOaT, out_a, out_a, 512,
                                        512, 512, 512,
                                        0, 0, 0, 0, 0, 0, 1, 1.0f, 0);
    gemm_wmma<0><<<g, blk, 0, stream>>>(o_t, WOtT, out_t, out_t, 512,
                                        512, 512, 512,
                                        0, 0, 0, 0, 0, 0, 1, 1.0f, 0);
  }
}
